// MinRNN_18391049961857
// MI455X (gfx1250) — matrix operand order repaired and checked by reading, not compiled, not run on hardware
//
#include <hip/hip_runtime.h>
#include <hip/hip_bf16.h>

// ---------------------------------------------------------------------------
// minLSTM forward for MI455X (gfx1250, wave32, WMMA + async global->LDS).
//   B=64, S=1024, E=U=1024, V=32000.
//   Stage 1: transpose+convert Wf/Wi/Wh -> bf16 [U][E] (N-major)
//   Stage 2: fused gather + triple GEMM (bf16 WMMA, f32 accum); weights are
//            staged with global_load_async_to_lds_b128 (ASYNCcnt), activations
//            via VGPR + v_cvt_pk_bf16_f32; 2-phase double-buffered pipeline.
//   Stage 3: sequential scan over S (HBM streaming, float4)
//   Stage 4: tiny MLP head -> sigmoid -> [B] f32
// ---------------------------------------------------------------------------

typedef __attribute__((ext_vector_type(8)))  __bf16 v8bf;
typedef __attribute__((ext_vector_type(16))) __bf16 v16bf;
typedef __attribute__((ext_vector_type(8)))  float  v8f;

#define BDIM 64
#define SDIM 1024
#define EDIM 1024
#define UDIM 1024
#define MTOT (BDIM * SDIM)     // 65536 GEMM rows
#define KSTEP 32
#define LDA 40                 // padded LDS stride (halves) to spread banks

__device__ __forceinline__ v16bf ld_frag(const __bf16* lo, const __bf16* hi) {
    union { v8bf u[2]; v16bf v; } c;
    c.u[0] = *(const v8bf*)lo;
    c.u[1] = *(const v8bf*)hi;
    return c.v;
}

__device__ __forceinline__ v8bf cvt8(float4 a, float4 b) {
    v8bf p;
    p[0] = (__bf16)a.x; p[1] = (__bf16)a.y; p[2] = (__bf16)a.z; p[3] = (__bf16)a.w;
    p[4] = (__bf16)b.x; p[5] = (__bf16)b.y; p[6] = (__bf16)b.z; p[7] = (__bf16)b.w;
    return p;
}

// Async copy 16B global -> LDS (per lane). Generic-pointer low 32 bits are the
// wave-relative LDS address.
__device__ __forceinline__ void async_w_load(const __bf16* src, __bf16* ldsdst) {
    asm volatile("global_load_async_to_lds_b128 %0, %1, off"
                 :: "v"((unsigned)(uintptr_t)ldsdst),
                    "v"((unsigned long long)(uintptr_t)src)
                 : "memory");
}
__device__ __forceinline__ void wait_async() {
    asm volatile("s_wait_asynccnt 0x0" ::: "memory");
}

// ---------------------------------------------------------------------------
// Stage 1: W [E][U] f32 -> Wt [3][U][E] bf16 (N-major), LDS-staged transpose
// ---------------------------------------------------------------------------
__global__ __launch_bounds__(1024)
void transpose_convert(const float* __restrict__ Wf,
                       const float* __restrict__ Wi,
                       const float* __restrict__ Wh,
                       __bf16* __restrict__ Wt) {
    __shared__ float tile[32][33];
    const float* W = (blockIdx.z == 0) ? Wf : (blockIdx.z == 1) ? Wi : Wh;
    const int n0 = blockIdx.x * 32, k0 = blockIdx.y * 32;
    const int tx = threadIdx.x, ty = threadIdx.y;
    tile[ty][tx] = W[(size_t)(k0 + ty) * UDIM + n0 + tx];
    __syncthreads();
    Wt[((size_t)blockIdx.z * UDIM + n0 + ty) * EDIM + k0 + tx] = (__bf16)tile[tx][ty];
}

// ---------------------------------------------------------------------------
// Stage 2: fused gather + 3x GEMM (bf16 WMMA) + gate epilogue.
// Block: 256 threads = 8 waves. Tile: M=64, N=64. Wave w -> M-subtile w&3,
// N-subtiles {2*(w>>2), 2*(w>>2)+1}, 3 gates => 6 v8f accumulators.
// K loop unrolled x2 so each phase uses a compile-time LDS buffer.
// ---------------------------------------------------------------------------
__global__ __launch_bounds__(256)
void gate_gemm(const int*    __restrict__ sentence,
               const float*  __restrict__ emb,
               const __bf16* __restrict__ Wt,       // [3][U][E] bf16
               const float*  __restrict__ bf_,
               const float*  __restrict__ bi_,
               const float*  __restrict__ bh_,
               float* __restrict__ fn_out,          // [M][U]
               float* __restrict__ ih_out) {        // [M][U]
    __shared__ __align__(16) __bf16 ldsA[2][64 * LDA];
    __shared__ __align__(16) __bf16 ldsB[2][3][64 * LDA];
    __shared__ int ldsIdx[64];

    const int tid  = threadIdx.x;
    const int lane = tid & 31;
    const int wave = tid >> 5;
    const int m0   = blockIdx.y * 64;
    const int n0   = blockIdx.x * 64;

    if (tid < 64) ldsIdx[tid] = sentence[m0 + tid];

    const int msub  = wave & 3;
    const int npair = wave >> 2;

    const int arow = tid >> 2;          // 0..63: LDS fill row
    const int acol = (tid & 3) * 8;     // 0,8,16,24: 8 halves per thread

    __syncthreads();                    // ldsIdx visible

    const size_t embRow = (size_t)ldsIdx[arow] * EDIM + acol;
    size_t wRow[3];
#pragma unroll
    for (int g = 0; g < 3; ++g)
        wRow[g] = ((size_t)g * UDIM + n0 + arow) * EDIM + acol;

    // ---- prologue: stage tile k=0 into buffer 0 ----
    {
        const float* src = emb + embRow;
        float4 f0 = *(const float4*)(src);
        float4 f1 = *(const float4*)(src + 4);
#pragma unroll
        for (int g = 0; g < 3; ++g)
            async_w_load(Wt + wRow[g], &ldsB[0][g][arow * LDA + acol]);
        *(v8bf*)&ldsA[0][arow * LDA + acol] = cvt8(f0, f1);
        wait_async();
    }
    __syncthreads();

    v8f acc[2][3];
#pragma unroll
    for (int nt = 0; nt < 2; ++nt)
#pragma unroll
        for (int g = 0; g < 3; ++g)
            acc[nt][g] = (v8f){0.f, 0.f, 0.f, 0.f, 0.f, 0.f, 0.f, 0.f};

    const int arf  = msub * 16 + (lane & 15);
    const int quad = lane >> 4;
    const int bcol = lane & 15;

    // One pipeline phase: consume LDS buffer {Abuf,Bbuf}, prefetch column kk
    // into {AbufN,BbufN} (weights async straight to LDS, A via VGPR+cvt).
    auto phase = [&](const __bf16* Abuf, const __bf16 (*Bbuf)[64 * LDA],
                     __bf16* AbufN, __bf16 (*BbufN)[64 * LDA], int kk) {
        const bool more = kk < EDIM;
        float4 pf0, pf1;
        if (more) {
#pragma unroll
            for (int g = 0; g < 3; ++g)
                async_w_load(Wt + wRow[g] + kk, &BbufN[g][arow * LDA + acol]);
            const float* src = emb + embRow + kk;
            pf0 = *(const float4*)(src);
            pf1 = *(const float4*)(src + 4);
        }

        const v16bf afrag = ld_frag(&Abuf[arf * LDA + quad * 8],
                                    &Abuf[arf * LDA + 16 + quad * 8]);
        v16bf bfr[2][3];
#pragma unroll
        for (int nt = 0; nt < 2; ++nt) {
            const int bbase = ((npair * 2 + nt) * 16 + bcol) * LDA + quad * 8;
#pragma unroll
            for (int g = 0; g < 3; ++g)
                bfr[nt][g] = ld_frag(&Bbuf[g][bbase], &Bbuf[g][bbase + 16]);
        }
#pragma unroll
        for (int nt = 0; nt < 2; ++nt)
#pragma unroll
            for (int g = 0; g < 3; ++g)
                acc[nt][g] = __builtin_amdgcn_wmma_f32_16x16x32_bf16(
                    false, afrag, false, bfr[nt][g], (short)0, acc[nt][g],
                    false, false);

        if (more)
            *(v8bf*)&AbufN[arow * LDA + acol] = cvt8(pf0, pf1);
        wait_async();          // async weights landed in BbufN
        __syncthreads();
    };

    for (int k0 = 0; k0 < EDIM; k0 += 2 * KSTEP) {
        phase(ldsA[0], ldsB[0], ldsA[1], ldsB[1], k0 + KSTEP);
        phase(ldsA[1], ldsB[1], ldsA[0], ldsB[0], k0 + 2 * KSTEP);
    }

    // ---- epilogue: bias + sigmoid + normalize, store f_n and i_n*h_tilde ----
    const int quadrow = quad * 8;   // C/D layout: M = r + 8*(lane>=16)
#pragma unroll
    for (int nt = 0; nt < 2; ++nt) {
        const int ncol = n0 + (npair * 2 + nt) * 16 + bcol;
        const float bF = bf_[ncol], bI = bi_[ncol], bH = bh_[ncol];
#pragma unroll
        for (int r = 0; r < 8; ++r) {
            const float fv = acc[nt][0][r] + bF;
            const float iv = acc[nt][1][r] + bI;
            const float hv = acc[nt][2][r] + bH;
            const float sf  = __builtin_amdgcn_rcpf(1.f + __expf(-fv));
            const float si  = __builtin_amdgcn_rcpf(1.f + __expf(-iv));
            const float inv = __builtin_amdgcn_rcpf(sf + si);
            const int m = m0 + msub * 16 + quadrow + r;
            const size_t off = (size_t)m * UDIM + ncol;
            fn_out[off] = sf * inv;
            ih_out[off] = si * inv * hv;
        }
    }
}

// ---------------------------------------------------------------------------
// Stage 3: h[b,u] = scan over s of h = f*h + ih.  One thread per (b, 4 u's).
// ---------------------------------------------------------------------------
__global__ __launch_bounds__(256)
void scan_kernel(const float4* __restrict__ fn,
                 const float4* __restrict__ ih,
                 float4* __restrict__ h) {
    const int t  = blockIdx.x * blockDim.x + threadIdx.x;   // 0..16383
    const int b  = t >> 8;
    const int c4 = t & 255;                                 // float4 column
    size_t base = (size_t)b * SDIM * (UDIM / 4) + c4;
    float4 hv = {0.f, 0.f, 0.f, 0.f};
#pragma unroll 4
    for (int s = 0; s < SDIM; ++s) {
        const float4 f = fn[base];
        const float4 g = ih[base];
        hv.x = fmaf(f.x, hv.x, g.x);
        hv.y = fmaf(f.y, hv.y, g.y);
        hv.z = fmaf(f.z, hv.z, g.z);
        hv.w = fmaf(f.w, hv.w, g.w);
        base += UDIM / 4;
    }
    h[t] = hv;
}

// ---------------------------------------------------------------------------
// Stage 4: out[b] = sigmoid((h[b,:] @ W1 + b1) @ W2 + b2)
// ---------------------------------------------------------------------------
__global__ __launch_bounds__(64)
void head_kernel(const float* __restrict__ h,
                 const float* __restrict__ W1, const float* __restrict__ b1,
                 const float* __restrict__ W2, const float* __restrict__ b2,
                 float* __restrict__ out) {
    __shared__ float red[64];
    const int b = blockIdx.x, j = threadIdx.x;
    const float* hr = h + (size_t)b * UDIM;
    float acc = b1[j];
    for (int k = 0; k < UDIM; ++k)
        acc = fmaf(hr[k], W1[k * 64 + j], acc);
    red[j] = acc * W2[j];
    __syncthreads();
#pragma unroll
    for (int off = 32; off > 0; off >>= 1) {
        if (j < off) red[j] += red[j + off];
        __syncthreads();
    }
    if (j == 0)
        out[b] = __builtin_amdgcn_rcpf(1.f + __expf(-(red[0] + b2[0])));
}

// ---------------------------------------------------------------------------
extern "C" void kernel_launch(void* const* d_in, const int* in_sizes, int n_in,
                              void* d_out, int out_size, void* d_ws, size_t ws_size,
                              hipStream_t stream) {
    const int*   sentence = (const int*)  d_in[0];
    const float* emb      = (const float*)d_in[1];
    const float* Wf       = (const float*)d_in[2];
    const float* bf_      = (const float*)d_in[3];
    const float* Wi       = (const float*)d_in[4];
    const float* bi_      = (const float*)d_in[5];
    const float* Wh       = (const float*)d_in[6];
    const float* bh_      = (const float*)d_in[7];
    const float* W1       = (const float*)d_in[8];
    const float* b1       = (const float*)d_in[9];
    const float* W2       = (const float*)d_in[10];
    const float* b2       = (const float*)d_in[11];
    float* out = (float*)d_out;

    char* ws = (char*)d_ws;
    const size_t wt_bytes = (size_t)3 * UDIM * EDIM * sizeof(__bf16);   // 6 MB
    const size_t gm_bytes = (size_t)MTOT * UDIM * sizeof(float);        // 256 MB
    __bf16* Wt = (__bf16*)ws;
    float* fn = (float*)(ws + wt_bytes);
    float* ih = (float*)(ws + wt_bytes + gm_bytes);
    float* h  = (float*)(ws + wt_bytes + 2 * gm_bytes);

    transpose_convert<<<dim3(UDIM / 32, EDIM / 32, 3), dim3(32, 32), 0, stream>>>(
        Wf, Wi, Wh, Wt);

    gate_gemm<<<dim3(UDIM / 64, MTOT / 64), 256, 0, stream>>>(
        sentence, emb, Wt, bf_, bi_, bh_, fn, ih);

    scan_kernel<<<dim3((BDIM * UDIM / 4) / 256), 256, 0, stream>>>(
        (const float4*)fn, (const float4*)ih, (float4*)h);

    head_kernel<<<dim3(BDIM), 64, 0, stream>>>(h, W1, b1, W2, b2, out);
}
